// SparseResidualBlock_37288906063940
// MI455X (gfx1250) — matrix operand order repaired and checked by reading, not compiled, not run on hardware
//
#include <hip/hip_runtime.h>

typedef __attribute__((ext_vector_type(16))) __bf16 v16bf;
typedef __attribute__((ext_vector_type(8)))  float  v8f;

static constexpr int CDIM = 128;          // channels in == out
static constexpr float BN_EPS = 1e-4f;
static constexpr int ROWS_PER_TILE  = 64;
static constexpr int TILES_PER_BLOCK = 4;

// ---------- helpers ----------
__device__ __forceinline__ unsigned short f32_to_bf16(float f) {
  unsigned int u = __float_as_uint(f);
  u += 0x7FFFu + ((u >> 16) & 1u);        // round-to-nearest-even
  return (unsigned short)(u >> 16);
}

__device__ __forceinline__ void wait_async0() {
#if __has_builtin(__builtin_amdgcn_s_wait_asynccnt)
  __builtin_amdgcn_s_wait_asynccnt(0);
#else
  asm volatile("s_wait_asynccnt 0x0" ::: "memory");
#endif
}

// async gather of one 16-byte granule: global (L2) -> LDS, no VGPR round-trip
__device__ __forceinline__ void async_gather_b128(unsigned lds_off,
                                                  const void* gaddr) {
  asm volatile("global_load_async_to_lds_b128 %0, %1, off"
               :: "v"(lds_off), "v"((unsigned long long)(uintptr_t)gaddr)
               : "memory");
}

// ---------- utility kernels ----------
__global__ void zero_f32(float* __restrict__ p, long n) {
  long i = (long)blockIdx.x * blockDim.x + threadIdx.x;
  long stride = (long)gridDim.x * blockDim.x;
  for (; i < n; i += stride) p[i] = 0.f;
}

// features fp32 [N,128] -> bf16 [(N+1),128], dummy row N = 0
__global__ void convert_feats(const float* __restrict__ f,
                              unsigned short* __restrict__ o, int N) {
  long n = (long)(N + 1) * CDIM;
  long i = (long)blockIdx.x * blockDim.x + threadIdx.x;
  long stride = (long)gridDim.x * blockDim.x;
  for (; i < n; i += stride) {
    long r = i >> 7;
    o[i] = (r < N) ? f32_to_bf16(f[i]) : (unsigned short)0;
  }
}

// W fp32 [27,128,128] (cin,cout) -> bf16 B-fragment layout:
//   [off 27][kc 4][cb 8][lane 32][h 16]
//   lane 0-15 : K = kc*32 + h        (N-col = cb*16 + lane)
//   lane 16-31: K = kc*32 + 16 + h   (N-col = cb*16 + lane-16)
__global__ void convert_w(const float* __restrict__ W,
                          unsigned short* __restrict__ wf) {
  int i = blockIdx.x * blockDim.x + threadIdx.x;
  if (i >= 27 * 4 * 8 * 32 * 16) return;
  int h    = i & 15;
  int lane = (i >> 4) & 31;
  int cb   = (i >> 9) & 7;
  int kc   = (i >> 12) & 3;
  int off  = i >> 14;
  int cin  = kc * 32 + ((lane >> 4) << 4) + h;
  int cout = cb * 16 + (lane & 15);
  wf[i] = f32_to_bf16(W[((long)off * CDIM + cin) * CDIM + cout]);
}

// ---------- sparse-conv GEMM via WMMA, async double-buffered gather ----------
// grid.x = groups of 4 row-tiles (64 rows each), grid.y = 27 offsets,
// 256 threads = 8 waves; wave w owns output columns [w*16, w*16+16).
__global__ __launch_bounds__(256) void subm_conv_wmma(
    const unsigned short* __restrict__ featsA,  // (N+1) x 128 bf16
    const unsigned short* __restrict__ wf,      // B-fragment layout
    const int* __restrict__ in_idx,             // [27, N] (dummy = N)
    const int* __restrict__ out_idx,            // [27, N]
    float* __restrict__ acc,                    // (N+1) x 128 f32
    int N)
{
  __shared__ __align__(16) unsigned short As[2][ROWS_PER_TILE * CDIM]; // 2x16KB

  const int  tid   = threadIdx.x;
  const int  off   = blockIdx.y;
  const long tile0 = (long)blockIdx.x * TILES_PER_BLOCK;

  const int lane  = tid & 31;
  const int cb    = tid >> 5;
  const int lhalf = lane >> 4;
  const int col   = (cb << 4) + (lane & 15);

  // B fragments: invariant over the 4 tiles -> load once per block
  v16bf b[4];
#pragma unroll
  for (int kc = 0; kc < 4; ++kc) {
    const int4* bp = (const int4*)(wf +
        ((((size_t)off * 4 + kc) * 8 + cb) * 32 + lane) * 16);
    union { v16bf v; int4 q[2]; } u;
    u.q[0] = bp[0];
    u.q[1] = bp[1];
    b[kc] = u.v;
  }

  // async gather of one 64-row tile into LDS buffer `buf`
  auto issue_gather = [&](int t, int buf) {
    const long brow = (tile0 + t) * ROWS_PER_TILE;
#pragma unroll
    for (int i = tid; i < ROWS_PER_TILE * 16; i += 256) {  // 16 int4 per row
      int  row  = i >> 4, seg = i & 15;
      long grow = brow + row;
      int  src  = (grow < N) ? in_idx[(size_t)off * N + grow] : N;
      unsigned lds_off =
          (unsigned)(uintptr_t)&(((int4*)As[buf])[row * 16 + seg]);
      async_gather_b128(lds_off,
                        ((const int4*)featsA) + ((size_t)src * 16 + seg));
    }
  };

  issue_gather(0, 0);

  for (int t = 0; t < TILES_PER_BLOCK; ++t) {
    const int buf = t & 1;
    wait_async0();          // this wave's async gathers done
    __syncthreads();        // all waves' gathers visible in LDS

    if (t + 1 < TILES_PER_BLOCK)
      issue_gather(t + 1, buf ^ 1);   // overlap next gather with compute

    const long brow = (tile0 + t) * ROWS_PER_TILE;
    const unsigned short* A = As[buf];

#pragma unroll
    for (int rt = 0; rt < 4; ++rt) {
      v8f c = {};
      const int arow = rt * 16 + (lane & 15);
#pragma unroll
      for (int kc = 0; kc < 4; ++kc) {
        // contiguous A k order used here (same as B packing): lanes 0-15 take K = {0..15}+kc*32,
        //                  lanes 16-31 take K = {16..31}+kc*32
        const int4* ap = (const int4*)&A[arow * CDIM + kc * 32 + (lhalf << 4)];
        union { v16bf v; int4 q[2]; } a;
        a.q[0] = ap[0];        // 8 halves
        a.q[1] = ap[1];        // +8 halves
        c = __builtin_amdgcn_wmma_f32_16x16x32_bf16(
                false, a.v, false, b[kc], (short)0, c, false, false);
      }
      // scatter-add D tile: VGPR g -> row M = g + 8*lhalf, col = lane&15
#pragma unroll
      for (int g = 0; g < 8; ++g) {
        long grow = brow + rt * 16 + g + (lhalf << 3);
        int  orow = (grow < N) ? out_idx[(size_t)off * N + grow] : N;
        atomicAdd(&acc[(size_t)orow * CDIM + col], c[g]);
      }
    }
    __syncthreads();        // all reads of As[buf] done before it is reused
  }
}

// ---------- batch-norm ----------
__global__ void bn_stats(const float* __restrict__ x,
                         float* __restrict__ stats, int N) {
  int col     = threadIdx.x & (CDIM - 1);
  int slice   = blockIdx.x * (blockDim.x / CDIM) + (threadIdx.x / CDIM);
  int nslices = gridDim.x * (blockDim.x / CDIM);
  float s = 0.f, ss = 0.f;
  for (long r = slice; r < N; r += nslices) {
    float v = x[r * CDIM + col];
    s += v; ss += v * v;
  }
  atomicAdd(&stats[col], s);
  atomicAdd(&stats[CDIM + col], ss);
}

// BN -> ReLU -> bf16 (feeds second conv); dummy row N = 0
__global__ void bn_relu_to_bf16(const float* __restrict__ x,
                                const float* __restrict__ stats,
                                const float* __restrict__ gamma,
                                const float* __restrict__ beta,
                                unsigned short* __restrict__ o, int N) {
  long  n    = (long)(N + 1) * CDIM;
  long  i    = (long)blockIdx.x * blockDim.x + threadIdx.x;
  long  strd = (long)gridDim.x * blockDim.x;
  float invN = 1.0f / (float)N;
  for (; i < n; i += strd) {
    int  col = (int)(i & (CDIM - 1));
    long r   = i >> 7;
    float y  = 0.f;
    if (r < N) {
      float m  = stats[col] * invN;
      float vv = stats[CDIM + col] * invN - m * m;
      y = (x[i] - m) * rsqrtf(vv + BN_EPS) * gamma[col] + beta[col];
      y = fmaxf(y, 0.f);
    }
    o[i] = f32_to_bf16(y);
  }
}

// BN2 + residual + ReLU -> fp32 output
__global__ void bn_residual_relu(const float* __restrict__ x,
                                 const float* __restrict__ stats,
                                 const float* __restrict__ gamma,
                                 const float* __restrict__ beta,
                                 const float* __restrict__ resid,
                                 float* __restrict__ out, int N) {
  long  n    = (long)N * CDIM;
  long  i    = (long)blockIdx.x * blockDim.x + threadIdx.x;
  long  strd = (long)gridDim.x * blockDim.x;
  float invN = 1.0f / (float)N;
  for (; i < n; i += strd) {
    int   col = (int)(i & (CDIM - 1));
    float m   = stats[col] * invN;
    float vv  = stats[CDIM + col] * invN - m * m;
    float y = (x[i] - m) * rsqrtf(vv + BN_EPS) * gamma[col] + beta[col]
              + resid[i];
    out[i] = fmaxf(y, 0.f);
  }
}

// ---------- host launcher ----------
extern "C" void kernel_launch(void* const* d_in, const int* in_sizes, int n_in,
                              void* d_out, int out_size, void* d_ws, size_t ws_size,
                              hipStream_t stream) {
  const float* features = (const float*)d_in[0];
  const float* W1       = (const float*)d_in[1];
  const float* gamma1   = (const float*)d_in[2];
  const float* beta1    = (const float*)d_in[3];
  const float* W2       = (const float*)d_in[4];
  const float* gamma2   = (const float*)d_in[5];
  const float* beta2    = (const float*)d_in[6];
  const int*   in_idx   = (const int*)d_in[7];
  const int*   out_idx  = (const int*)d_in[8];

  const int N = in_sizes[0] / CDIM;

  // workspace carve-up (256B aligned regions)
  char*  w          = (char*)d_ws;
  size_t featsBytes = (((size_t)(N + 1) * CDIM * 2) + 255) & ~(size_t)255;
  size_t accBytes   = (((size_t)(N + 1) * CDIM * 4) + 255) & ~(size_t)255;
  size_t wfBytes    = ((size_t)(27 * 4 * 8 * 32 * 16 * 2) + 255) & ~(size_t)255;

  unsigned short* featsA = (unsigned short*)w;  w += featsBytes;
  unsigned short* feats1 = (unsigned short*)w;  w += featsBytes;
  float*          acc    = (float*)w;           w += accBytes;
  unsigned short* wf1    = (unsigned short*)w;  w += wfBytes;
  unsigned short* wf2    = (unsigned short*)w;  w += wfBytes;
  float*          stats  = (float*)w;           // 512 floats (2 BN stat sets)

  const long accN   = (long)(N + 1) * CDIM;
  const int  nrb    = (N + ROWS_PER_TILE - 1) / ROWS_PER_TILE;
  const int  ngrp   = (nrb + TILES_PER_BLOCK - 1) / TILES_PER_BLOCK;
  const dim3 convG(ngrp, 27);
  const int  wElems = 27 * 4 * 8 * 32 * 16;

  // prep: converts + zero accumulator + zero stats
  convert_feats<<<2048, 256, 0, stream>>>(features, featsA, N);
  convert_w<<<(wElems + 255) / 256, 256, 0, stream>>>(W1, wf1);
  convert_w<<<(wElems + 255) / 256, 256, 0, stream>>>(W2, wf2);
  zero_f32<<<2048, 256, 0, stream>>>(acc, accN);
  zero_f32<<<2, 256, 0, stream>>>(stats, 512);

  // conv1 -> BN1 -> ReLU -> bf16
  subm_conv_wmma<<<convG, 256, 0, stream>>>(featsA, wf1, in_idx, out_idx, acc, N);
  bn_stats<<<256, 256, 0, stream>>>(acc, stats, N);
  bn_relu_to_bf16<<<2048, 256, 0, stream>>>(acc, stats, gamma1, beta1, feats1, N);

  // conv2 (reuse accumulator) -> BN2 -> +residual -> ReLU
  zero_f32<<<2048, 256, 0, stream>>>(acc, accN);
  subm_conv_wmma<<<convG, 256, 0, stream>>>(feats1, wf2, in_idx, out_idx, acc, N);
  bn_stats<<<256, 256, 0, stream>>>(acc, stats + 256, N);
  bn_residual_relu<<<2048, 256, 0, stream>>>(acc, stats + 256, gamma2, beta2,
                                             features, (float*)d_out, N);
}
